// CSMHP_6004364280423
// MI455X (gfx1250) — compile-verified, matrix operand order checked by reading, not run on hardware
//
#include <hip/hip_runtime.h>

typedef __attribute__((ext_vector_type(16))) _Float16 v16h;
typedef __attribute__((ext_vector_type(8)))  float    v8f;

#define TWIN  100.0f
#define MAXN  4096
#define LOG2E 1.4426950408889634f

static __device__ __forceinline__ float fast_exp2(float x) {
#if __has_builtin(__builtin_amdgcn_exp2f)
    return __builtin_amdgcn_exp2f(x);   // raw v_exp_f32
#else
    return exp2f(x);
#endif
}

// Kernel 1: per 16-row i-tile, E[c,i] = sum_{j<i} exp(-beta_c (t_i - t_j))
// via f16 WMMA row-sum (B = ones). 256 threads = 8 waves; wave w -> cluster w.
// Main loop is unmasked (all j < i0 <= i); only the diagonal tile is masked.
__global__ __launch_bounds__(256) void hawkes_excite_ll_kernel(
    const float* __restrict__ P,  const float* __restrict__ et,
    const float* __restrict__ mu, const float* __restrict__ ga,
    const float* __restrict__ al, const float* __restrict__ be,
    float* __restrict__ part, int N)
{
    __shared__ alignas(16) float sh_t[MAXN];
    __shared__ float sh_E[8][16];
    __shared__ float sh_red[16];

    const int tid = threadIdx.x;
    for (int k = tid; k < MAXN; k += 256)
        sh_t[k] = (k < N) ? et[k] : 0.0f;
    __syncthreads();

    const int lane = tid & 31;
    const int c    = tid >> 5;          // cluster id (C == 8)
    const int half = lane >> 4;
    const int M    = lane & 15;
    const int i0   = (int)blockIdx.x * 16;
    const int i    = i0 + M;
    const float ti = sh_t[(i < MAXN) ? i : 0];

    // exp(-beta*(ti - tj)) = exp2(s*tj - s*ti),  s = beta*log2(e)
    const float s  = be[c] * LOG2E;
    const float c0 = -s * ti;

    v16h ones;
    #pragma unroll
    for (int e = 0; e < 16; ++e) ones[e] = (_Float16)1.0f;

    const float4* st4 = (const float4*)sh_t;
    v8f acc = {};

    // ---- full tiles: j0 + 32 <= i0, no masking needed ----
    const int jfull = i0 & ~31;
    for (int j0 = 0; j0 < jfull; j0 += 32) {
        const int b4 = (j0 >> 2) + (half << 1);
        const float4 q0 = st4[b4];
        const float4 q1 = st4[b4 + 1];
        const float4 q2 = st4[b4 + 4];
        const float4 q3 = st4[b4 + 5];
        const float tj[16] = { q0.x, q0.y, q0.z, q0.w,  q1.x, q1.y, q1.z, q1.w,
                               q2.x, q2.y, q2.z, q2.w,  q3.x, q3.y, q3.z, q3.w };
        v16h a;
        #pragma unroll
        for (int e = 0; e < 16; ++e)
            a[e] = (_Float16)fast_exp2(__builtin_fmaf(s, tj[e], c0));
        acc = __builtin_amdgcn_wmma_f32_16x16x32_f16(false, a, false, ones,
                                                     (short)0, acc, false, false);
    }

    // ---- diagonal tile: mask j < i per element ----
    {
        const int j0 = jfull;
        const int b4 = (j0 >> 2) + (half << 1);
        const float4 q0 = st4[b4];
        const float4 q1 = st4[b4 + 1];
        const float4 q2 = st4[b4 + 4];
        const float4 q3 = st4[b4 + 5];
        const float tj[16] = { q0.x, q0.y, q0.z, q0.w,  q1.x, q1.y, q1.z, q1.w,
                               q2.x, q2.y, q2.z, q2.w,  q3.x, q3.y, q3.z, q3.w };
        v16h a;
        #pragma unroll
        for (int e = 0; e < 16; ++e) {
            // K(e,l) = (e/8)*16 + (l/16)*8 + (e%8)
            const int K = ((e >> 3) << 4) | (half << 3) | (e & 7);
            const int j = j0 + K;
            const float v = (j < i) ? fast_exp2(__builtin_fmaf(s, tj[e], c0)) : 0.0f;
            a[e] = (_Float16)v;
        }
        acc = __builtin_amdgcn_wmma_f32_16x16x32_f16(false, a, false, ones,
                                                     (short)0, acc, false, false);
    }

    // D columns are identical; lane with n==0 of each half writes its 8 rows.
    if ((lane & 15) == 0) {
        #pragma unroll
        for (int r = 0; r < 8; ++r)
            sh_E[c][r + 8 * half] = acc[r];
    }
    __syncthreads();

    if (tid < 16) {
        const int ii = i0 + tid;
        float lg = 0.0f;
        if (ii < N) {
            const float t_i = sh_t[ii];
            float inten = 0.0f;
            #pragma unroll
            for (int cc = 0; cc < 8; ++cc) {
                const float lam = al[cc] * sh_E[cc][tid] + mu[cc]
                                + ga[cc] * (t_i * (1.0f / TWIN));
                inten += lam * P[ii * 8 + cc];
            }
            lg = __logf(inten);
        }
        sh_red[tid] = lg;
    }
    __syncthreads();
    if (tid == 0) {
        float sum = 0.0f;
        #pragma unroll
        for (int r = 0; r < 16; ++r) sum += sh_red[r];
        part[blockIdx.x] = sum;
    }
}

// Kernel 2: single block. Integral term + deterministic final combine.
__global__ __launch_bounds__(256) void hawkes_final_kernel(
    const float* __restrict__ P,  const float* __restrict__ et,
    const float* __restrict__ mu, const float* __restrict__ ga,
    const float* __restrict__ al, const float* __restrict__ be,
    const float* __restrict__ part, float* __restrict__ out,
    int N, int nblk)
{
    __shared__ float sS2[8][257];
    __shared__ float sS1[8][257];
    const int tid = threadIdx.x;
    const float tl = et[N - 1];

    float b[8], p2[8], p1[8];
    #pragma unroll
    for (int c = 0; c < 8; ++c) { b[c] = be[c]; p2[c] = 0.0f; p1[c] = 0.0f; }

    // S2[c] = sum_{j=0}^{N-2} exp(-beta_c (t_last - t_j))
    for (int j = tid; j < N - 1; j += 256) {
        const float d = -(tl - et[j]) * LOG2E;
        #pragma unroll
        for (int c = 0; c < 8; ++c) p2[c] += fast_exp2(b[c] * d);
    }
    // S1[c] = sum_i P[i,c]
    for (int i = tid; i < N; i += 256) {
        #pragma unroll
        for (int c = 0; c < 8; ++c) p1[c] += P[i * 8 + c];
    }
    #pragma unroll
    for (int c = 0; c < 8; ++c) { sS2[c][tid] = p2[c]; sS1[c][tid] = p1[c]; }
    __syncthreads();

    if (tid == 0) {
        float ll = 0.0f;
        for (int bix = 0; bix < nblk; ++bix) ll += part[bix];

        const float t0    = et[0];
        const float tdiff = tl - t0;
        const float tsq   = tl * tl - t0 * t0;
        float integral = 0.0f;
        for (int c = 0; c < 8; ++c) {
            float S2 = 0.0f, S1 = 0.0f;
            for (int k = 0; k < 256; ++k) { S2 += sS2[c][k]; S1 += sS1[c][k]; }
            const float ab      = al[c] / b[c];
            const float expterm = ab * (float)(N - 1) - ab * S2;
            const float basetrm = tdiff * mu[c] + tsq * ga[c] * (1.0f / (2.0f * TWIN));
            integral += S1 * (expterm + basetrm);
        }
        integral /= (float)N;
        out[0] = integral - ll;   // -(ll_sum - integral_part)
    }
}

extern "C" void kernel_launch(void* const* d_in, const int* in_sizes, int n_in,
                              void* d_out, int out_size, void* d_ws, size_t ws_size,
                              hipStream_t stream) {
    const float* P  = (const float*)d_in[0];  // (N, C) row-major
    const float* et = (const float*)d_in[1];  // (N,)
    const float* mu = (const float*)d_in[2];  // (C,)
    const float* ga = (const float*)d_in[3];  // (C,)
    const float* al = (const float*)d_in[4];  // (C,)
    const float* be = (const float*)d_in[5];  // (C,)
    const int N = in_sizes[1];                // 4096
    float* out  = (float*)d_out;
    float* part = (float*)d_ws;               // nblk partial log-sums

    const int nblk = (N + 15) / 16;
    hawkes_excite_ll_kernel<<<nblk, 256, 0, stream>>>(P, et, mu, ga, al, be, part, N);
    hawkes_final_kernel<<<1, 256, 0, stream>>>(P, et, mu, ga, al, be, part, out, N, nblk);
}
